// PPFEncoder_46926812676441
// MI455X (gfx1250) — matrix operand order repaired and checked by reading, not compiled, not run on hardware
//
#include <hip/hip_runtime.h>
#include <hip/hip_bf16.h>

#define NPTS 1024
#define EPSV 1e-7f

typedef _Float16 v16h __attribute__((ext_vector_type(16)));
typedef _Float16 v8h  __attribute__((ext_vector_type(8)));
typedef float    v8f  __attribute__((ext_vector_type(8)));

#define AROW 40   // halves per staged A row -> 80B row stride (16B aligned)
#define CROW 33   // floats per staged C row

__device__ __forceinline__ void lds_fence() {
    asm volatile("s_wait_dscnt 0" ::: "memory");
}

__device__ __forceinline__ v8f wmma_f16(v16h a, v16h b, v8f c) {
    return __builtin_amdgcn_wmma_f32_16x16x32_f16(false, a, false, b, (short)0, c,
                                                  false, false);
}

__device__ __forceinline__ v8f splat8(float v) {
    v8f c;
#pragma unroll
    for (int r = 0; r < 8; ++r) c[r] = v;
    return c;
}

__device__ __forceinline__ v8f addsplat8(v8f x, float v) {
    v8f c;
#pragma unroll
    for (int r = 0; r < 8; ++r) c[r] = x[r] + v;
    return c;
}

// Build a B-fragment (KxN f16, N-column = ncol) from a row-major f32 weight
// matrix W[k*dout + n]. Layout: lanes 0-15 hold K={0..7,16..23},
// lanes 16-31 hold K={8..15,24..31}, column = lane%16; element e <-> K=kbase+e (e<8), 16+kbase+e-8 (e>=8).
// k >= kdim or ncol < 0 -> zero fill (K / N padding).
__device__ __forceinline__ v16h load_bfrag(const float* __restrict__ W, int dout,
                                           int kdim, int ncol) {
    const int lane  = threadIdx.x & 31;
    const int kbase = (lane >> 4) << 3;
    v16h b;
#pragma unroll
    for (int e = 0; e < 16; ++e) {
        const int k = (e < 8) ? kbase + e : 16 + kbase + (e - 8);
        const float w = (ncol >= 0 && k < kdim) ? W[k * dout + ncol] : 0.0f;
        b[e] = (_Float16)w;
    }
    return b;
}

__device__ __forceinline__ v16h combine16(v8h lo, v8h hi) {
    return __builtin_shufflevector(lo, hi, 0, 1, 2, 3, 4, 5, 6, 7,
                                   8, 9, 10, 11, 12, 13, 14, 15);
}

// A-fragment (16x32 f16) from LDS row-major tile. ISA layout:
// lane<16: row=lane,   K = {0..7, 16..23}
// lane>=16: row=lane-16, K = {8..15, 24..31}
__device__ __forceinline__ v16h load_afrag32(const _Float16* __restrict__ sa) {
    const int lane = threadIdx.x & 31;
    const int row  = lane & 15;
    const int kb   = (lane >> 4) << 3;  // 0 or 8
    const v8h lo = *(const v8h*)(sa + row * AROW + kb);
    const v8h hi = *(const v8h*)(sa + row * AROW + kb + 16);
    return combine16(lo, hi);
}

// A-fragment for K=16 matrices, zero padded to K=32.
__device__ __forceinline__ v16h load_afrag16(const _Float16* __restrict__ sa) {
    const int lane = threadIdx.x & 31;
    const int row  = lane & 15;
    const int kb   = (lane >> 4) << 3;
    const v8h lo = *(const v8h*)(sa + row * AROW + kb);
    const v8h z  = {};
    return combine16(lo, z);
}

// Store a C/D fragment (f32, ISA C-layout) into the f16 LDS staging tile.
__device__ __forceinline__ void stage_d(_Float16* __restrict__ sa, v8f d,
                                        int ncolbase, bool relu) {
    const int lane  = threadIdx.x & 31;
    const int col   = ncolbase + (lane & 15);
    const int rbase = (lane >> 4) << 3;  // 0 or 8
#pragma unroll
    for (int r = 0; r < 8; ++r) {
        float v = d[r];
        if (relu) v = fmaxf(v, 0.0f);
        sa[(rbase + r) * AROW + col] = (_Float16)v;
    }
}

// Stage 1: per-point partial products of layer-0 weights against features.
// A0/A1 fold the biases; B0/B1 are the f_j halves.
__global__ __launch_bounds__(256) void ppf_pre_kernel(
    const float* __restrict__ feat, const float* __restrict__ w0,
    const float* __restrict__ b0, const float* __restrict__ w1,
    const float* __restrict__ b1, float* __restrict__ A0,
    float* __restrict__ B0, float* __restrict__ A1, float* __restrict__ B1) {
    const int t  = blockIdx.x * blockDim.x + threadIdx.x;  // 1024*32 threads
    const int pt = t >> 5;
    const int n  = t & 31;
    const float* f = feat + pt * 40;
    float a0 = b0[n], bb0 = 0.0f, a1 = b1[n], bb1 = 0.0f;
#pragma unroll 4
    for (int d = 0; d < 40; ++d) {
        const float fv = f[d];
        a0  += fv * w0[d * 32 + n];
        bb0 += fv * w0[(40 + d) * 32 + n];
        a1  += fv * w1[d * 32 + n];
        bb1 += fv * w1[(40 + d) * 32 + n];
    }
    A0[t] = a0; B0[t] = bb0; A1[t] = a1; B1[t] = bb1;
}

// Stage 2: one wave = one 16-pair tile (fixed i, j0..j0+15). Whole residual
// MLP runs through v_wmma_f32_16x16x32_f16 with LDS fragment staging.
__global__ __launch_bounds__(256) void ppf_main_kernel(
    const float* __restrict__ pc, const float* __restrict__ nrm,
    const float* __restrict__ dist,
    const float* __restrict__ A0, const float* __restrict__ B0,
    const float* __restrict__ A1, const float* __restrict__ B1,
    const float* __restrict__ W0, const float* __restrict__ W1,
    const float* __restrict__ l0w2, const float* __restrict__ l0b2,
    const float* __restrict__ l1w1, const float* __restrict__ l1b1,
    const float* __restrict__ l1w2, const float* __restrict__ l1b2,
    const float* __restrict__ l2w0, const float* __restrict__ l2b0,
    const float* __restrict__ l2w1, const float* __restrict__ l2b1,
    const float* __restrict__ l2w2, const float* __restrict__ l2b2,
    const float* __restrict__ fw, const float* __restrict__ fb,
    float* __restrict__ out) {
    __shared__ __align__(16) _Float16 s_a[8][16 * AROW];
    __shared__ float s_c[8][16 * CROW];
    __shared__ float s_ppf[8][16][4];

    const int wid   = threadIdx.x >> 5;
    const int lane  = threadIdx.x & 31;
    const int tile  = blockIdx.x * 8 + wid;
    const int i     = tile >> 6;
    const int j0    = (tile & 63) << 4;
    const int colLo = lane & 15;
    const int hi    = lane >> 4;
    const int rbase = hi * 8;

    _Float16* sa = s_a[wid];
    float*    sc = s_c[wid];

    // ---- resident B fragments (weights are tiny & L2-hot) ----
    const v16h B_l0fc2_0 = load_bfrag(l0w2, 32, 32, colLo);
    const v16h B_l0fc2_1 = load_bfrag(l0w2, 32, 32, 16 + colLo);
    const v16h B_l1fc1_0 = load_bfrag(l1w1, 32, 32, colLo);
    const v16h B_l1fc1_1 = load_bfrag(l1w1, 32, 32, 16 + colLo);
    const v16h B_l1fc2_0 = load_bfrag(l1w2, 32, 32, colLo);
    const v16h B_l1fc2_1 = load_bfrag(l1w2, 32, 32, 16 + colLo);
    const v16h B_l2fc0   = load_bfrag(l2w0, 16, 32, colLo);
    const v16h B_l2fc1   = load_bfrag(l2w1, 16, 32, colLo);
    const v16h B_l2fc2   = load_bfrag(l2w2, 16, 16, colLo);
    v16h B_fin[5];
#pragma unroll
    for (int nh = 0; nh < 5; ++nh) {
        const int nc = nh * 16 + colLo;
        B_fin[nh] = load_bfrag(fw, 66, 16, nc < 66 ? nc : -1);
    }

    // ---- ppf features for the 16 pairs of this tile ----
    const float pix = pc[i * 3 + 0], piy = pc[i * 3 + 1], piz = pc[i * 3 + 2];
    const float nix = nrm[i * 3 + 0], niy = nrm[i * 3 + 1], niz = nrm[i * 3 + 2];
    if (lane < 16) {
        const int j = j0 + lane;
        const float dx = pix - pc[j * 3 + 0];
        const float dy = piy - pc[j * 3 + 1];
        const float dz = piz - pc[j * 3 + 2];
        const float njx = nrm[j * 3 + 0], njy = nrm[j * 3 + 1],
                    njz = nrm[j * 3 + 2];
        const float d   = dist[i * NPTS + j];
        const float inv = 1.0f / (d + EPSV);
        s_ppf[wid][lane][0] = (nix * dx + niy * dy + niz * dz) * inv;
        s_ppf[wid][lane][1] = (njx * dx + njy * dy + njz * dz) * inv;
        s_ppf[wid][lane][2] = nix * njx + niy * njy + niz * njz;
        s_ppf[wid][lane][3] = d;
    }
    lds_fence();

    // ---- layer-0 pre-activations via rank-split (A_i + B_j + ppf@W[80:84]) ----
    {
        const int p = colLo;          // pair handled by this lane
        const int j = j0 + p;
        const float p0 = s_ppf[wid][p][0], p1 = s_ppf[wid][p][1];
        const float p2 = s_ppf[wid][p][2], p3 = s_ppf[wid][p][3];
        const int n0 = hi * 16;
#pragma unroll
        for (int t = 0; t < 16; ++t) {
            const int n = n0 + t;
            const float u0 = A0[i * 32 + n] + B0[j * 32 + n] +
                             p0 * W0[80 * 32 + n] + p1 * W0[81 * 32 + n] +
                             p2 * W0[82 * 32 + n] + p3 * W0[83 * 32 + n];
            float u1 = A1[i * 32 + n] + B1[j * 32 + n] +
                       p0 * W1[80 * 32 + n] + p1 * W1[81 * 32 + n] +
                       p2 * W1[82 * 32 + n] + p3 * W1[83 * 32 + n];
            sc[p * CROW + n] = u0;                      // residual (f32, C layout src)
            sa[p * AROW + n] = (_Float16)fmaxf(u1, 0.0f);  // relu branch -> A tile
        }
    }
    lds_fence();

    // ---- layer 0 fc2: x = relu(h) @ W2 + b2 + u0 ----
    v16h a = load_afrag32(sa);
    v8f c0, c1;
#pragma unroll
    for (int r = 0; r < 8; ++r) {
        c0[r] = sc[(rbase + r) * CROW + colLo] + l0b2[colLo];
        c1[r] = sc[(rbase + r) * CROW + 16 + colLo] + l0b2[16 + colLo];
    }
    v8f x0 = wmma_f16(a, B_l0fc2_0, c0);
    v8f x1 = wmma_f16(a, B_l0fc2_1, c1);

    // ---- layer 1 (32 -> 32, identity residual) ----
    stage_d(sa, x0, 0, false);
    stage_d(sa, x1, 16, false);
    lds_fence();
    a = load_afrag32(sa);
    v8f h0 = wmma_f16(a, B_l1fc1_0, splat8(l1b1[colLo]));
    v8f h1 = wmma_f16(a, B_l1fc1_1, splat8(l1b1[16 + colLo]));
    stage_d(sa, h0, 0, true);
    stage_d(sa, h1, 16, true);
    lds_fence();
    a  = load_afrag32(sa);
    x0 = wmma_f16(a, B_l1fc2_0, addsplat8(x0, l1b2[colLo]));
    x1 = wmma_f16(a, B_l1fc2_1, addsplat8(x1, l1b2[16 + colLo]));

    // ---- layer 2 (32 -> 16, projected residual) ----
    stage_d(sa, x0, 0, false);
    stage_d(sa, x1, 16, false);
    lds_fence();
    a = load_afrag32(sa);
    v8f xr = wmma_f16(a, B_l2fc0, splat8(l2b0[colLo]));
    v8f hh = wmma_f16(a, B_l2fc1, splat8(l2b1[colLo]));
    stage_d(sa, hh, 0, true);
    lds_fence();
    v16h a16 = load_afrag16(sa);              // K padded 16 -> 32
    v8f x2 = wmma_f16(a16, B_l2fc2, addsplat8(xr, l2b2[colLo]));

    // ---- final 16 -> 66 projection + store ----
    stage_d(sa, x2, 0, false);
    lds_fence();
    a16 = load_afrag16(sa);
    const long long obase = (long long)i * NPTS + j0;
#pragma unroll
    for (int nh = 0; nh < 5; ++nh) {
        const int n = nh * 16 + colLo;
        const float bv = (n < 66) ? fb[n] : 0.0f;
        const v8f dfin = wmma_f16(a16, B_fin[nh], splat8(bv));
        if (n < 66) {
#pragma unroll
            for (int r = 0; r < 8; ++r) {
                out[(obase + rbase + r) * 66 + n] = dfin[r];
            }
        }
    }
}

extern "C" void kernel_launch(void* const* d_in, const int* in_sizes, int n_in,
                              void* d_out, int out_size, void* d_ws,
                              size_t ws_size, hipStream_t stream) {
    const float* pc   = (const float*)d_in[0];
    const float* nrm  = (const float*)d_in[1];
    const float* dist = (const float*)d_in[2];
    const float* feat = (const float*)d_in[3];
    const float* l0w0 = (const float*)d_in[4];
    const float* l0b0 = (const float*)d_in[5];
    const float* l0w1 = (const float*)d_in[6];
    const float* l0b1 = (const float*)d_in[7];
    const float* l0w2 = (const float*)d_in[8];
    const float* l0b2 = (const float*)d_in[9];
    const float* l1w1 = (const float*)d_in[10];
    const float* l1b1 = (const float*)d_in[11];
    const float* l1w2 = (const float*)d_in[12];
    const float* l1b2 = (const float*)d_in[13];
    const float* l2w0 = (const float*)d_in[14];
    const float* l2b0 = (const float*)d_in[15];
    const float* l2w1 = (const float*)d_in[16];
    const float* l2b1 = (const float*)d_in[17];
    const float* l2w2 = (const float*)d_in[18];
    const float* l2b2 = (const float*)d_in[19];
    const float* fw   = (const float*)d_in[20];
    const float* fb   = (const float*)d_in[21];
    float* out = (float*)d_out;

    float* ws = (float*)d_ws;          // 4 x 1024 x 32 f32 = 512 KB
    float* A0 = ws;
    float* B0 = ws + 32768;
    float* A1 = ws + 65536;
    float* B1 = ws + 98304;

    ppf_pre_kernel<<<128, 256, 0, stream>>>(feat, l0w0, l0b0, l0w1, l0b1,
                                            A0, B0, A1, B1);
    // 1024 i-rows x 64 j-tiles = 65536 tiles; 8 waves/block -> 8192 blocks.
    ppf_main_kernel<<<8192, 256, 0, stream>>>(
        pc, nrm, dist, A0, B0, A1, B1, l0w0, l0w1, l0w2, l0b2, l1w1, l1b1,
        l1w2, l1b2, l2w0, l2b0, l2w1, l2b1, l2w2, l2b2, fw, fb, out);
}
